// OriginAttention_64467459113131
// MI455X (gfx1250) — matrix operand order repaired and checked by reading, not compiled, not run on hardware
//
#include <hip/hip_runtime.h>
#include <hip/hip_bf16.h>

typedef __bf16 bf16_t;
typedef __attribute__((ext_vector_type(16))) __bf16 v16bf;
typedef __attribute__((ext_vector_type(8)))  __bf16 v8bf;
typedef __attribute__((ext_vector_type(8)))  float   v8f;

#define HEADS  4
#define DHEAD  64
#define NB     4096   // h*w
#define CIN    128    // dim
#define OQKV   768    // 3*HEADS*DHEAD
#define INNER  256    // HEADS*DHEAD
#define BATCH  2
#define ROWT   2      // 16-row tiles per wave in attention (32 rows/wave)

// ---- fragment helpers (CDNA5 wave32 WMMA bf16 layouts, CDNA5 ISA §7.12.2) ----
__device__ __forceinline__ v16bf frag_cat(const bf16_t* p0, const bf16_t* p1) {
  v8bf a = *(const v8bf*)p0;
  v8bf b = *(const v8bf*)p1;
  v16bf r;
#pragma unroll
  for (int i = 0; i < 8; ++i) { r[i] = a[i]; r[i + 8] = b[i]; }
  return r;
}
// A 16x32 (MxK): lane row = lane&15; elem 0..7 -> K = 8*half.., elem 8..15 -> K = 16+8*half..
__device__ __forceinline__ v16bf load_a_frag(const bf16_t* rowptr, int half) {
  return frag_cat(rowptr + 8 * half, rowptr + 16 + 8 * half);
}
// B 32x16 (KxN): lane col = lane&15; elem 0..7 -> K = 8*half.., elem 8..15 -> K = 16+8*half..
__device__ __forceinline__ v16bf load_b_frag(const bf16_t* colptr, int half) {
  return frag_cat(colptr + 8 * half, colptr + 16 + 8 * half);
}

// ---- conversion / staging kernels ----
__global__ void OA_cvt_bf16(const float* __restrict__ src, bf16_t* __restrict__ dst, int n) {
  int i = blockIdx.x * blockDim.x + threadIdx.x;
  if (i < n) dst[i] = (bf16_t)src[i];
}

// xT[b][n][c] = bf16(x[b][c][n])  — makes B fragments for QKV GEMM contiguous per lane
__global__ void OA_transpose_x(const float* __restrict__ x, bf16_t* __restrict__ xT) {
  int i = blockIdx.x * blockDim.x + threadIdx.x;
  if (i >= BATCH * CIN * NB) return;
  int n = i % NB;
  int c = (i / NB) % CIN;
  int b = i / (NB * CIN);
  xT[((size_t)b * NB + n) * CIN + c] = (bf16_t)x[i];
}

// ---- QKV projection: C[o, n] = sum_c wqkv[o,c] * x[b,c,n]  (K=128, 4 WMMA steps) ----
// writes q,k: [b][h][n][64] (bf16), v transposed: [b][h][64][n] (bf16)
__global__ void OA_qkv_gemm(const bf16_t* __restrict__ wqkv, const bf16_t* __restrict__ xT,
                            bf16_t* __restrict__ qb, bf16_t* __restrict__ kb,
                            bf16_t* __restrict__ vTb) {
  const int lane = threadIdx.x & 31;
  const int wave = threadIdx.x >> 5;
  const int half = lane >> 4;
  const int l15  = lane & 15;
  const int n_base = (blockIdx.x * 8 + wave) * 16;
  const int o_base = blockIdx.y * 16;
  const int b      = blockIdx.z;

  const bf16_t* arow = wqkv + (size_t)(o_base + l15) * CIN;
  const bf16_t* brow = xT + ((size_t)b * NB + n_base + l15) * CIN;

  v8f acc = {};
#pragma unroll
  for (int kk = 0; kk < CIN; kk += 32) {
    v16bf a  = load_a_frag(arow + kk, half);
    v16bf bb = load_b_frag(brow + kk, half);
    acc = __builtin_amdgcn_wmma_f32_16x16x32_bf16(false, a, false, bb, (short)0, acc, false, false);
  }

  const int n = n_base + l15;
#pragma unroll
  for (int r = 0; r < 8; ++r) {
    int o    = o_base + r + 8 * half;     // D layout: M = r + 8*half, N = lane&15
    int t    = o >> 8;                    // 0=q 1=k 2=v
    int rem  = o & 255;
    int head = rem >> 6;
    int d    = rem & 63;
    bf16_t val = (bf16_t)acc[r];
    size_t bh = (size_t)b * HEADS + head;
    if (t == 0)      qb [(bh * NB + n) * DHEAD + d] = val;
    else if (t == 1) kb [(bh * NB + n) * DHEAD + d] = val;
    else             vTb[(bh * DHEAD + d) * NB + n] = val;
  }
}

// ---- attention: O[n,d] = sigmoid(Q K^T * 1/8) @ V per (b,head) ----
// 4 waves/block, each wave owns a 32-row Q strip (2 x 16-row WMMA tiles) so each
// K/V fragment fetched from L2 feeds 2 WMMAs instead of 1. Streams m in steps of 32.
// Writes attT[b][pos][c'] replicating reference's transpose(0,2,1,3)+reshape permutation:
//   flat = n*256 + head*64 + d ; c' = n>>4 ; pos = (n&15)*256 + head*64 + d
__global__ void OA_attention(const bf16_t* __restrict__ q, const bf16_t* __restrict__ k,
                             const bf16_t* __restrict__ vT, bf16_t* __restrict__ attT) {
  const int lane = threadIdx.x & 31;
  const int wave = threadIdx.x >> 5;
  const int half = lane >> 4;
  const int l15  = lane & 15;
  const int bh   = blockIdx.y;          // b*HEADS + head
  const int head = bh & (HEADS - 1);
  const int b    = bh >> 2;
  const int n_base = blockIdx.x * (4 * ROWT * 16) + wave * (ROWT * 16);

  const bf16_t* qh = q  + (size_t)bh * NB * DHEAD;
  const bf16_t* kh = k  + (size_t)bh * NB * DHEAD;
  const bf16_t* vh = vT + (size_t)bh * DHEAD * NB;

  __shared__ bf16_t lds_p[4][ROWT][16][32];   // per-wave P tiles (D-layout -> A-layout bounce)

  v16bf a_q[ROWT][2];
#pragma unroll
  for (int rt = 0; rt < ROWT; ++rt)
#pragma unroll
    for (int dk = 0; dk < 2; ++dk)
      a_q[rt][dk] = load_a_frag(qh + (size_t)(n_base + rt * 16 + l15) * DHEAD + dk * 32, half);

  v8f o_acc[ROWT][4] = {};
  const float scale = 0.125f;           // DIM_HEAD^-0.5

  for (int mt = 0; mt < NB; mt += 32) {
    // K fragments for this m-step (shared by both row tiles)
    v16bf bk[2][2];
#pragma unroll
    for (int mc = 0; mc < 2; ++mc)
#pragma unroll
      for (int dk = 0; dk < 2; ++dk)
        bk[mc][dk] = load_b_frag(kh + (size_t)(mt + mc * 16 + l15) * DHEAD + dk * 32, half);

    // prefetch next m-step's K/V lines into cache while WMMAs run
    if (mt + 32 < NB) {
      __builtin_prefetch(kh + (size_t)(mt + 32 + l15) * DHEAD, 0, 3);
      __builtin_prefetch(vh + (size_t)l15 * NB + mt + 32, 0, 3);
    }

    // S = Q K^T tiles, sigmoid -> LDS as bf16
#pragma unroll
    for (int rt = 0; rt < ROWT; ++rt) {
#pragma unroll
      for (int mc = 0; mc < 2; ++mc) {
        v8f s = {};
#pragma unroll
        for (int dk = 0; dk < 2; ++dk)
          s = __builtin_amdgcn_wmma_f32_16x16x32_bf16(false, a_q[rt][dk], false, bk[mc][dk],
                                                      (short)0, s, false, false);
#pragma unroll
        for (int e = 0; e < 8; ++e) {
          float xv = s[e] * scale;
          float pv = 1.0f / (1.0f + __expf(-xv));
          lds_p[wave][rt][e + 8 * half][mc * 16 + l15] = (bf16_t)pv;   // [M][m]
        }
      }
    }
    __builtin_amdgcn_wave_barrier();    // keep store->load order (per-wave LDS is in-order in HW)
    v16bf a_p[ROWT];
#pragma unroll
    for (int rt = 0; rt < ROWT; ++rt)
      a_p[rt] = frag_cat(&lds_p[wave][rt][l15][8 * half], &lds_p[wave][rt][l15][16 + 8 * half]);
    __builtin_amdgcn_wave_barrier();

    // O += P @ V  (V fragments shared by both row tiles)
#pragma unroll
    for (int j = 0; j < 4; ++j) {
      v16bf bv = load_b_frag(vh + (size_t)(j * 16 + l15) * NB + mt, half);
#pragma unroll
      for (int rt = 0; rt < ROWT; ++rt)
        o_acc[rt][j] = __builtin_amdgcn_wmma_f32_16x16x32_bf16(false, a_p[rt], false, bv,
                                                               (short)0, o_acc[rt][j], false, false);
    }
  }

#pragma unroll
  for (int rt = 0; rt < ROWT; ++rt) {
#pragma unroll
    for (int j = 0; j < 4; ++j) {
#pragma unroll
      for (int r = 0; r < 8; ++r) {
        int ni  = n_base + rt * 16 + r + 8 * half;
        int di  = j * 16 + l15;
        int cc  = head * DHEAD + di;
        int pos = (ni & 15) * INNER + cc;
        int cp  = ni >> 4;
        attT[((size_t)b * NB + pos) * INNER + cp] = (bf16_t)o_acc[rt][j][r];
      }
    }
  }
}

// ---- output projection: out[b][o][pos] = sum_c' wout[o][c'] * attT[b][pos][c'] + b_out[o] ----
__global__ void OA_out_gemm(const bf16_t* __restrict__ wout, const bf16_t* __restrict__ attT,
                            const float* __restrict__ b_out, float* __restrict__ out) {
  const int lane = threadIdx.x & 31;
  const int wave = threadIdx.x >> 5;
  const int half = lane >> 4;
  const int l15  = lane & 15;
  const int pos_base = (blockIdx.x * 8 + wave) * 16;
  const int o_base   = blockIdx.y * 16;
  const int b        = blockIdx.z;

  const bf16_t* arow = wout + (size_t)(o_base + l15) * INNER;
  const bf16_t* brow = attT + ((size_t)b * NB + pos_base + l15) * INNER;

  v8f acc = {};
#pragma unroll
  for (int kk = 0; kk < INNER; kk += 32) {
    v16bf a  = load_a_frag(arow + kk, half);
    v16bf bb = load_b_frag(brow + kk, half);
    acc = __builtin_amdgcn_wmma_f32_16x16x32_bf16(false, a, false, bb, (short)0, acc, false, false);
  }
  const int pos = pos_base + l15;
#pragma unroll
  for (int r = 0; r < 8; ++r) {
    int o = o_base + r + 8 * half;
    out[((size_t)b * CIN + o) * NB + pos] = acc[r] + b_out[o];
  }
}

extern "C" void kernel_launch(void* const* d_in, const int* in_sizes, int n_in,
                              void* d_out, int out_size, void* d_ws, size_t ws_size,
                              hipStream_t stream) {
  const float* x     = (const float*)d_in[0];   // (2,128,64,64)
  const float* w_qkv = (const float*)d_in[1];   // (768,128)
  const float* w_out = (const float*)d_in[2];   // (128,256)
  const float* b_out = (const float*)d_in[3];   // (128,)
  float* out = (float*)d_out;                   // (2,128,64,64)

  char* ws = (char*)d_ws;
  size_t off = 0;
  auto carve = [&](size_t bytes) { char* p = ws + off; off += (bytes + 255) & ~(size_t)255; return p; };
  bf16_t* xT    = (bf16_t*)carve((size_t)BATCH * NB * CIN * sizeof(bf16_t));           // 2 MB
  bf16_t* wqkvb = (bf16_t*)carve((size_t)OQKV * CIN * sizeof(bf16_t));
  bf16_t* woutb = (bf16_t*)carve((size_t)CIN * INNER * sizeof(bf16_t));
  bf16_t* qb    = (bf16_t*)carve((size_t)BATCH * HEADS * NB * DHEAD * sizeof(bf16_t)); // 4 MB
  bf16_t* kb    = (bf16_t*)carve((size_t)BATCH * HEADS * NB * DHEAD * sizeof(bf16_t)); // 4 MB
  bf16_t* vTb   = (bf16_t*)carve((size_t)BATCH * HEADS * DHEAD * NB * sizeof(bf16_t)); // 4 MB
  bf16_t* attTb = (bf16_t*)carve((size_t)BATCH * NB * INNER * sizeof(bf16_t));         // 4 MB
  (void)ws_size; (void)n_in; (void)in_sizes; (void)out_size;

  OA_transpose_x<<<(BATCH * CIN * NB + 255) / 256, 256, 0, stream>>>(x, xT);
  OA_cvt_bf16<<<(OQKV * CIN + 255) / 256, 256, 0, stream>>>(w_qkv, wqkvb, OQKV * CIN);
  OA_cvt_bf16<<<(CIN * INNER + 255) / 256, 256, 0, stream>>>(w_out, woutb, CIN * INNER);

  // 48 o-tiles x (32 groups x 8 waves) n-tiles x 2 batches
  OA_qkv_gemm<<<dim3(32, 48, 2), 256, 0, stream>>>(wqkvb, xT, qb, kb, vTb);

  // 32 row-blocks (4 waves x 32 rows) x 8 (b*heads)
  OA_attention<<<dim3(NB / (4 * ROWT * 16), BATCH * HEADS), 128, 0, stream>>>(qb, kb, vTb, attTb);

  // 8 o-tiles x (32 groups x 8 waves) pos-tiles x 2 batches
  OA_out_gemm<<<dim3(32, 8, 2), 256, 0, stream>>>(woutb, attTb, b_out, out);
}
